// TransformerEncoderLayer_88433376624798
// MI455X (gfx1250) — matrix operand order repaired and checked by reading, not compiled, not run on hardware
//
#include <hip/hip_runtime.h>
#include <math.h>

// ---------------- types ----------------
typedef __attribute__((ext_vector_type(16))) _Float16 v16h;
typedef __attribute__((ext_vector_type(8)))  float    v8f;

#define S_LEN 4096
#define DMODEL 768
#define NHEAD 12
#define HDIM 64
#define FFDIM 3072

// ---------------- async global -> LDS (CDNA5, ASYNCcnt tracked) ----------------
// the VDST operand holds the LDS address; generic LDS addr truncated to 32 bits is the
// LDS byte offset (ISA 10.2 aperture mapping discards upper bits).
__device__ __forceinline__ void async_ld_b128(_Float16* lds, const _Float16* g) {
  unsigned int       l  = (unsigned int)(unsigned long long)(uintptr_t)lds;
  unsigned long long ga = (unsigned long long)(uintptr_t)g;
  asm volatile("global_load_async_to_lds_b128 %0, %1, off"
               :: "v"(l), "v"(ga) : "memory");
}
#define WAIT_ASYNC(n) asm volatile("s_wait_asynccnt " #n ::: "memory")

// Load 16 halves as a WMMA fragment: two 16B chunks at p and p+second_off halves.
__device__ __forceinline__ v16h ldfrag(const _Float16* p, int second_off) {
  union { uint4 u[2]; v16h v; } t;
  t.u[0] = *(const uint4*)(p);
  t.u[1] = *(const uint4*)(p + second_off);
  return t.v;
}

// A fragment (16x32 f16): lane<16 -> K {kk..kk+7, kk+16..kk+23}; lane>=16 -> +8.
__device__ __forceinline__ v16h ldfragA(const _Float16* base, int row, int stride,
                                        int kk, int lane) {
  return ldfrag(base + (size_t)row * stride + kk + ((lane & 16) ? 8 : 0), 16);
}
// B fragment (32x16 f16): lane<16 -> K {kk..kk+7, kk+16..kk+23}; lane>=16 -> +8.
__device__ __forceinline__ v16h ldfragB(const _Float16* base, int col, int stride,
                                        int kk, int lane) {
  return ldfrag(base + (size_t)col * stride + kk + ((lane & 16) ? 8 : 0), 16);
}

__device__ __forceinline__ v8f wmma_f16(v16h a, v16h b, v8f c) {
  return __builtin_amdgcn_wmma_f32_16x16x32_f16(false, a, false, b, (short)0, c,
                                                false, false);
}

// ---------------- weight conversion ----------------
// wq (H, D, HD) f32 -> WT [H*HD][D] f16  (B-transposed layout [N][K])
__global__ __launch_bounds__(256) void cvt_qkv(const float* __restrict__ w,
                                               _Float16* __restrict__ wt) {
  int idx = blockIdx.x * 256 + threadIdx.x;   // over 768*768
  int d = idx % DMODEL;
  int j = idx / DMODEL;                       // j = h*64 + k
  int h = j >> 6, k = j & 63;
  wt[idx] = (_Float16)w[((size_t)h * DMODEL + d) * HDIM + k];
}

// w [K][N] f32 -> wt [N][K] f16
__global__ __launch_bounds__(256) void cvt_T(const float* __restrict__ w,
                                             _Float16* __restrict__ wt,
                                             int Kd, int Nd) {
  int idx = blockIdx.x * 256 + threadIdx.x;
  if (idx >= Kd * Nd) return;
  int kk = idx % Kd, n = idx / Kd;
  wt[idx] = (_Float16)w[(size_t)kk * Nd + n];
}

// ---------------- LayerNorm (D=768, one row per block of 256) ----------------
__global__ __launch_bounds__(256) void ln_kernel(const float* __restrict__ x,
                                                 const float* __restrict__ g,
                                                 const float* __restrict__ b,
                                                 _Float16* __restrict__ out) {
  __shared__ float red[256];
  const int tid = threadIdx.x;
  const float* xr = x + (size_t)blockIdx.x * DMODEL;
  float v[3], s = 0.f;
#pragma unroll
  for (int j = 0; j < 3; ++j) { v[j] = xr[tid + j * 256]; s += v[j]; }
  red[tid] = s; __syncthreads();
  for (int off = 128; off; off >>= 1) {
    if (tid < off) red[tid] += red[tid + off];
    __syncthreads();
  }
  float mean = red[0] * (1.f / DMODEL);
  __syncthreads();
  float sq = 0.f;
#pragma unroll
  for (int j = 0; j < 3; ++j) { float d = v[j] - mean; sq += d * d; }
  red[tid] = sq; __syncthreads();
  for (int off = 128; off; off >>= 1) {
    if (tid < off) red[tid] += red[tid + off];
    __syncthreads();
  }
  float rs = rsqrtf(red[0] * (1.f / DMODEL) + 1e-5f);
  _Float16* o = out + (size_t)blockIdx.x * DMODEL;
#pragma unroll
  for (int j = 0; j < 3; ++j) {
    int i = tid + j * 256;
    o[i] = (_Float16)((v[j] - mean) * rs * g[i] + b[i]);
  }
}

// ---------------- WMMA GEMM: C[M,N] = A[M,K](f16) * BT[N,K](f16)^T ----------------
// 128x128 block tile, BK=32, 8 waves each 32x64, async double-buffered LDS.
// EPI 0: O16 = C + bias ; EPI 1: O16 = gelu(C + bias) ; EPI 2: O32 = resid + C + bias
#define BM 128
#define BN 128
#define BK 32
template <int EPI>
__global__ __launch_bounds__(256) void gemm_wmma(
    const _Float16* __restrict__ A, const _Float16* __restrict__ BT,
    const float* __restrict__ bias, const float* __restrict__ resid,
    _Float16* __restrict__ O16, float* __restrict__ O32, int M, int N, int K) {
  const int lane = threadIdx.x & 31;
  const int wid  = threadIdx.x >> 5;       // 8 waves
  const int wrow = wid >> 1;               // 0..3  (32-row strips)
  const int wcol = wid & 1;                // 0..1  (64-col strips)
  const int m0 = blockIdx.y * BM;
  const int n0 = blockIdx.x * BN;

  __shared__ _Float16 As[2][BM * BK];   // 2 x 8 KB
  __shared__ _Float16 Bs[2][BN * BK];   // 2 x 8 KB

  v8f acc[2][4] = {};

  // issue one tile (A + B) of async loads: 4 instructions per thread
  auto issue_tile = [&](int buf, int k0) {
    for (int i = threadIdx.x; i < BM * BK / 8; i += 256) {
      int r = i >> 2, seg = i & 3;   // BK/8 = 4 segments per row
      async_ld_b128(&As[buf][r * BK + seg * 8],
                    &A[(size_t)(m0 + r) * K + k0 + seg * 8]);
    }
    for (int i = threadIdx.x; i < BN * BK / 8; i += 256) {
      int r = i >> 2, seg = i & 3;
      async_ld_b128(&Bs[buf][r * BK + seg * 8],
                    &BT[(size_t)(n0 + r) * K + k0 + seg * 8]);
    }
  };

  issue_tile(0, 0);
  int buf = 0;
  for (int k0 = 0; k0 < K; k0 += BK, buf ^= 1) {
    if (k0 + BK < K) {
      issue_tile(buf ^ 1, k0 + BK);     // prefetch next tile (4 async ops)
      WAIT_ASYNC(4);                    // drain current tile (in-order)
    } else {
      WAIT_ASYNC(0);
    }
    __syncthreads();

    v16h af[2], bf[4];
#pragma unroll
    for (int i = 0; i < 2; ++i)
      af[i] = ldfragA(As[buf], wrow * 32 + i * 16 + (lane & 15), BK, 0, lane);
#pragma unroll
    for (int j = 0; j < 4; ++j)
      bf[j] = ldfragB(Bs[buf], wcol * 64 + j * 16 + (lane & 15), BK, 0, lane);
#pragma unroll
    for (int i = 0; i < 2; ++i)
#pragma unroll
      for (int j = 0; j < 4; ++j)
        acc[i][j] = wmma_f16(af[i], bf[j], acc[i][j]);
    __syncthreads();
  }

  const int hsel = (lane >> 4) & 1;
  const int nl = lane & 15;
#pragma unroll
  for (int i = 0; i < 2; ++i)
#pragma unroll
    for (int j = 0; j < 4; ++j)
#pragma unroll
      for (int r = 0; r < 8; ++r) {
        int m = m0 + wrow * 32 + i * 16 + r + hsel * 8;
        int n = n0 + wcol * 64 + j * 16 + nl;
        float c = acc[i][j][r] + bias[n];
        if (EPI == 0) {
          O16[(size_t)m * N + n] = (_Float16)c;
        } else if (EPI == 1) {
          O16[(size_t)m * N + n] =
              (_Float16)(0.5f * c * (1.f + erff(c * 0.70710678118654752f)));
        } else {
          O32[(size_t)m * N + n] = resid[(size_t)m * N + n] + c;
        }
      }
}

// ---------------- Flash attention (per head, 128 query rows / block) ----------------
__global__ __launch_bounds__(256) void attn_kernel(const _Float16* __restrict__ Q,
                                                   const _Float16* __restrict__ K,
                                                   const _Float16* __restrict__ V,
                                                   _Float16* __restrict__ O) {
  const int head = blockIdx.y;
  const int qblk = blockIdx.x;           // 0..31
  const int lane = threadIdx.x & 31;
  const int wid  = threadIdx.x >> 5;     // 8 waves, 16 query rows each
  const size_t hoff = (size_t)head * HDIM;

  __shared__ _Float16 Qs[128 * 64];      // 16 KB
  __shared__ _Float16 Ks[64 * 64];       //  8 KB
  __shared__ _Float16 VTs[64 * 64];      //  8 KB  (VT[d][t])
  __shared__ _Float16 Ps[8 * 16 * 64];   // 16 KB  (per-wave P tile)

  // async-load Q tile
  for (int i = threadIdx.x; i < 128 * 8; i += 256) {
    int r = i >> 3, seg = i & 7;
    async_ld_b128(&Qs[r * 64 + seg * 8],
                  &Q[(size_t)(qblk * 128 + r) * DMODEL + hoff + seg * 8]);
  }
  WAIT_ASYNC(0);
  __syncthreads();

  v16h qf[2];
#pragma unroll
  for (int kt = 0; kt < 2; ++kt)
    qf[kt] = ldfragA(Qs, wid * 16 + (lane & 15), 64, kt * 32, lane);

  v8f Oacc[4] = {};
  float mi[8], li[8];
#pragma unroll
  for (int r = 0; r < 8; ++r) { mi[r] = -1e30f; li[r] = 0.f; }

  for (int t0 = 0; t0 < S_LEN; t0 += 64) {
    // async K tile; V transposed via DS scalar stores
    for (int i = threadIdx.x; i < 64 * 8; i += 256) {
      int r = i >> 3, seg = i & 7;
      async_ld_b128(&Ks[r * 64 + seg * 8],
                    &K[(size_t)(t0 + r) * DMODEL + hoff + seg * 8]);
    }
    for (int i = threadIdx.x; i < 64 * 64; i += 256) {
      int t = i >> 6, d = i & 63;
      VTs[d * 64 + t] = V[(size_t)(t0 + t) * DMODEL + hoff + d];
    }
    WAIT_ASYNC(0);
    __syncthreads();

    // scores S[16x64] = Q @ K^T  (K tile used directly as B: B[d][t]=K[t][d])
    v8f sc[4] = {};
#pragma unroll
    for (int j = 0; j < 4; ++j)
#pragma unroll
      for (int kt = 0; kt < 2; ++kt) {
        v16h bf = ldfragB(Ks, j * 16 + (lane & 15), 64, kt * 32, lane);
        sc[j] = wmma_f16(qf[kt], bf, sc[j]);
      }
#pragma unroll
    for (int j = 0; j < 4; ++j)
#pragma unroll
      for (int r = 0; r < 8; ++r) sc[j][r] *= 0.125f;   // 1/sqrt(64)

    // online softmax, rows live in half-wave lane groups
#pragma unroll
    for (int r = 0; r < 8; ++r) {
      float cur = sc[0][r];
#pragma unroll
      for (int j = 1; j < 4; ++j) cur = fmaxf(cur, sc[j][r]);
#pragma unroll
      for (int msk = 1; msk < 16; msk <<= 1)
        cur = fmaxf(cur, __shfl_xor(cur, msk, 16));
      float mnew = fmaxf(mi[r], cur);
      float scale = __expf(mi[r] - mnew);
      float rsum = 0.f;
#pragma unroll
      for (int j = 0; j < 4; ++j) {
        float p = __expf(sc[j][r] - mnew);
        sc[j][r] = p;
        rsum += p;
      }
#pragma unroll
      for (int msk = 1; msk < 16; msk <<= 1)
        rsum += __shfl_xor(rsum, msk, 16);
      li[r] = li[r] * scale + rsum;
      mi[r] = mnew;
#pragma unroll
      for (int j = 0; j < 4; ++j) Oacc[j][r] *= scale;
    }

    // P -> per-wave LDS tile (A-source layout [16][64]), then P @ V
    _Float16* pw = &Ps[wid * 16 * 64];
    const int h8 = ((lane >> 4) & 1) * 8;
#pragma unroll
    for (int j = 0; j < 4; ++j)
#pragma unroll
      for (int r = 0; r < 8; ++r)
        pw[(r + h8) * 64 + j * 16 + (lane & 15)] = (_Float16)sc[j][r];
    asm volatile("s_wait_dscnt 0" ::: "memory");   // in-order per-wave DS RAW fence

    v16h pf[2];
#pragma unroll
    for (int kt = 0; kt < 2; ++kt)
      pf[kt] = ldfragA(pw, lane & 15, 64, kt * 32, lane);
#pragma unroll
    for (int j = 0; j < 4; ++j)
#pragma unroll
      for (int kt = 0; kt < 2; ++kt) {
        v16h vf = ldfragB(VTs, j * 16 + (lane & 15), 64, kt * 32, lane);
        Oacc[j] = wmma_f16(pf[kt], vf, Oacc[j]);
      }
    __syncthreads();
  }

  const int h8 = ((lane >> 4) & 1) * 8;
#pragma unroll
  for (int j = 0; j < 4; ++j)
#pragma unroll
    for (int r = 0; r < 8; ++r) {
      int m = qblk * 128 + wid * 16 + r + h8;
      O[(size_t)m * DMODEL + hoff + j * 16 + (lane & 15)] =
          (_Float16)(Oacc[j][r] / li[r]);
    }
}

// ---------------- launcher ----------------
extern "C" void kernel_launch(void* const* d_in, const int* in_sizes, int n_in,
                              void* d_out, int out_size, void* d_ws, size_t ws_size,
                              hipStream_t stream) {
  const float* x     = (const float*)d_in[0];
  const float* wq    = (const float*)d_in[1];
  const float* bq    = (const float*)d_in[2];
  const float* wk    = (const float*)d_in[3];
  const float* bk    = (const float*)d_in[4];
  const float* wv    = (const float*)d_in[5];
  const float* bv    = (const float*)d_in[6];
  const float* wo    = (const float*)d_in[7];
  const float* bo    = (const float*)d_in[8];
  const float* g1    = (const float*)d_in[9];
  const float* b1    = (const float*)d_in[10];
  const float* g2    = (const float*)d_in[11];
  const float* b2    = (const float*)d_in[12];
  const float* wff1  = (const float*)d_in[13];
  const float* bff1  = (const float*)d_in[14];
  const float* wff2  = (const float*)d_in[15];
  const float* bff2  = (const float*)d_in[16];
  float* out = (float*)d_out;

  char* ws = (char*)d_ws;
  size_t off = 0;
  auto alloc = [&](size_t bytes) -> void* {
    void* p = ws + off;
    off += (bytes + 255) & ~(size_t)255;
    return p;
  };
  const size_t SD = (size_t)S_LEN * DMODEL;
  _Float16* h16   = (_Float16*)alloc(SD * 2);
  _Float16* q16   = (_Float16*)alloc(SD * 2);
  _Float16* k16   = (_Float16*)alloc(SD * 2);
  _Float16* v16   = (_Float16*)alloc(SD * 2);
  _Float16* a16   = (_Float16*)alloc(SD * 2);
  float*    x1    = (float*)   alloc(SD * 4);
  _Float16* h216  = (_Float16*)alloc(SD * 2);
  _Float16* ff16  = (_Float16*)alloc((size_t)S_LEN * FFDIM * 2);
  _Float16* wqT   = (_Float16*)alloc((size_t)DMODEL * DMODEL * 2);
  _Float16* wkT   = (_Float16*)alloc((size_t)DMODEL * DMODEL * 2);
  _Float16* wvT   = (_Float16*)alloc((size_t)DMODEL * DMODEL * 2);
  _Float16* woT   = (_Float16*)alloc((size_t)DMODEL * DMODEL * 2);
  _Float16* wf1T  = (_Float16*)alloc((size_t)DMODEL * FFDIM * 2);
  _Float16* wf2T  = (_Float16*)alloc((size_t)DMODEL * FFDIM * 2);

  const int cblk = (DMODEL * DMODEL) / 256;            // 2304
  cvt_qkv<<<cblk, 256, 0, stream>>>(wq, wqT);
  cvt_qkv<<<cblk, 256, 0, stream>>>(wk, wkT);
  cvt_qkv<<<cblk, 256, 0, stream>>>(wv, wvT);
  cvt_T<<<cblk, 256, 0, stream>>>(wo, woT, DMODEL, DMODEL);
  cvt_T<<<(DMODEL * FFDIM) / 256, 256, 0, stream>>>(wff1, wf1T, DMODEL, FFDIM);
  cvt_T<<<(DMODEL * FFDIM) / 256, 256, 0, stream>>>(wff2, wf2T, FFDIM, DMODEL);

  // LN1 -> h16
  ln_kernel<<<S_LEN, 256, 0, stream>>>(x, g1, b1, h16);

  // Q/K/V projections
  dim3 gP(DMODEL / BN, S_LEN / BM);   // (6, 32)
  gemm_wmma<0><<<gP, 256, 0, stream>>>(h16, wqT, bq, nullptr, q16, nullptr,
                                       S_LEN, DMODEL, DMODEL);
  gemm_wmma<0><<<gP, 256, 0, stream>>>(h16, wkT, bk, nullptr, k16, nullptr,
                                       S_LEN, DMODEL, DMODEL);
  gemm_wmma<0><<<gP, 256, 0, stream>>>(h16, wvT, bv, nullptr, v16, nullptr,
                                       S_LEN, DMODEL, DMODEL);

  // Flash attention -> a16 [s][h*64+d]
  attn_kernel<<<dim3(S_LEN / 128, NHEAD), 256, 0, stream>>>(q16, k16, v16, a16);

  // x1 = x + attn @ wo + bo
  gemm_wmma<2><<<gP, 256, 0, stream>>>(a16, woT, bo, x, nullptr, x1,
                                       S_LEN, DMODEL, DMODEL);

  // LN2 -> h216
  ln_kernel<<<S_LEN, 256, 0, stream>>>(x1, g2, b2, h216);

  // ff = gelu(h2 @ w_ff1 + b_ff1)
  dim3 gF1(FFDIM / BN, S_LEN / BM);   // (24, 32)
  gemm_wmma<1><<<gF1, 256, 0, stream>>>(h216, wf1T, bff1, nullptr, ff16, nullptr,
                                        S_LEN, FFDIM, DMODEL);

  // out = x1 + ff @ w_ff2 + b_ff2
  gemm_wmma<2><<<gP, 256, 0, stream>>>(ff16, wf2T, bff2, x1, nullptr, out,
                                       S_LEN, DMODEL, FFDIM);
}
